// SignAwareAttention_1640677507228
// MI455X (gfx1250) — matrix operand order repaired and checked by reading, not compiled, not run on hardware
//
#include <hip/hip_runtime.h>
#include <cmath>

typedef unsigned short u16;
typedef unsigned int u32;
typedef __attribute__((ext_vector_type(16))) __bf16 v16bf;
typedef __attribute__((ext_vector_type(8)))  float  v8f;

union Frag { uint4 q[2]; v16bf v; };
union Acc  { v8f v; float f[8]; };

__device__ __forceinline__ u16 f2bf(float f) {
    u32 u = __float_as_uint(f);
    u32 r = u + 0x7FFFu + ((u >> 16) & 1u);   // round-to-nearest-even
    return (u16)(r >> 16);
}
__device__ __forceinline__ float bf2f(u16 h) {
    return __uint_as_float(((u32)h) << 16);
}

// ---------------------------------------------------------------------------
// Shared WMMA GEMM mainloop: C[BM=128 x BN=64] tile, BK=64 per stage.
// 256 threads = 8 waves; wave w owns 16-row band, 4 col-tiles (acc[4]).
// A staged row-major [128][72] (pad kills bank conflicts, keeps 16B align).
// B staged transposed  [64][72]  so B-fragments are contiguous b128 loads.
// bf16 A/B fragments follow CDNA5 16-bit WMMA VGPR layouts (ISA 7.12.2).
// ---------------------------------------------------------------------------
#define ASTR 72
#define BSTR 72

template<bool ABS_A>
__device__ __forceinline__ void gemm_core(
    const u16* __restrict__ A, int lda,
    const u16* __restrict__ B, int ldb,
    int m0, int n0, int K,
    u16* As, u16* Bs, v8f acc[4])
{
    const int tid  = threadIdx.x;
    const int wave = tid >> 5;
    const int lane = tid & 31;
    const int mrow = lane & 15;
    const int hf   = lane >> 4;

    Acc z;
#pragma unroll
    for (int j = 0; j < 8; j++) z.f[j] = 0.0f;
#pragma unroll
    for (int i = 0; i < 4; i++) acc[i] = z.v;

    for (int kb = 0; kb < K; kb += 64) {
        // ---- stage A tile: 128 rows x 64 cols (2 threads/row, 64B each) ----
        {
            const int r = tid >> 1, ho = tid & 1;
            const uint4* s = reinterpret_cast<const uint4*>(
                A + (size_t)(m0 + r) * lda + kb + ho * 32);
            uint4* d = reinterpret_cast<uint4*>(As + r * ASTR + ho * 32);
#pragma unroll
            for (int i = 0; i < 4; i++) {
                uint4 v = s[i];
                if (ABS_A) {   // |x|^p from sgn(x)|x|^p: clear bf16 sign bits
                    v.x &= 0x7FFF7FFFu; v.y &= 0x7FFF7FFFu;
                    v.z &= 0x7FFF7FFFu; v.w &= 0x7FFF7FFFu;
                }
                d[i] = v;
            }
        }
        // ---- stage B tile transposed: 64 k-rows x 64 n-cols -> Bs[n][k] ----
        {
            const int k = tid >> 2, seg = tid & 3;
            const u16* s = B + (size_t)(kb + k) * ldb + n0 + seg * 16;
            uint4 v0 = *reinterpret_cast<const uint4*>(s);
            uint4 v1 = *reinterpret_cast<const uint4*>(s + 8);
            const u16* e0 = reinterpret_cast<const u16*>(&v0);
            const u16* e1 = reinterpret_cast<const u16*>(&v1);
#pragma unroll
            for (int j = 0; j < 8; j++) Bs[(seg * 16 + j) * BSTR + k] = e0[j];
#pragma unroll
            for (int j = 0; j < 8; j++) Bs[(seg * 16 + 8 + j) * BSTR + k] = e1[j];
        }
        __syncthreads();

#pragma unroll
        for (int ks = 0; ks < 2; ks++) {
            // A frag: lane half hf -> K = {8hf..8hf+7, 16+8hf..23+8hf}
            Frag a;
            const u16* ar = As + (wave * 16 + mrow) * ASTR + ks * 32;
            a.q[0] = *reinterpret_cast<const uint4*>(ar + hf * 8);
            a.q[1] = *reinterpret_cast<const uint4*>(ar + 16 + hf * 8);
#pragma unroll
            for (int ct = 0; ct < 4; ct++) {
                // B frag: lane half hf -> K = {8hf..8hf+7, 16+8hf..23+8hf}, n=lane%16
                Frag b;
                const u16* br = Bs + (ct * 16 + mrow) * BSTR + ks * 32;
                b.q[0] = *reinterpret_cast<const uint4*>(br + hf * 8);
                b.q[1] = *reinterpret_cast<const uint4*>(br + 16 + hf * 8);
                acc[ct] = __builtin_amdgcn_wmma_f32_16x16x32_bf16(
                    false, a.v, false, b.v, (short)0, acc[ct], false, false);
            }
        }
        __syncthreads();
    }
}

// ---------------------------------------------------------------------------
// Setup kernels
// ---------------------------------------------------------------------------
__global__ __launch_bounds__(256) void cvt_f32_bf16(
    const float* __restrict__ s, u16* __restrict__ d, int n)
{
    int i = blockIdx.x * 256 + threadIdx.x;
    const int stride = gridDim.x * 256;
    for (; i < n; i += stride) d[i] = f2bf(s[i]);
}

__global__ __launch_bounds__(256) void pack_qkv(
    const float* __restrict__ Wq, const float* __restrict__ Wk,
    const float* __restrict__ Wvs, const float* __restrict__ Wvc,
    u16* __restrict__ W)
{
    const int i = blockIdx.x * 256 + threadIdx.x;   // 512*1536 exact
    if (i >= 512 * 1536) return;
    const int r = i / 1536, c = i % 1536;
    float v;
    if      (c <  512) v = Wq [r * 512 + c];
    else if (c < 1024) v = Wk [r * 512 + c - 512];
    else if (c < 1280) v = Wvs[r * 256 + c - 1024];
    else               v = Wvc[r * 256 + c - 1280];
    W[i] = f2bf(v);
}

__global__ __launch_bounds__(256) void pv_kernel(
    const float* __restrict__ power, float* __restrict__ pv)
{
    const int i = blockIdx.x * 256 + threadIdx.x;
    if (i < 512) pv[i] = 1.0f + 3.0f / (1.0f + expf(-power[i]));  // 1+ALPHA*sigmoid
}

// ---------------------------------------------------------------------------
// G1: fused projection GEMM  Y[32768 x 1536] = x_bf16 @ [Wq|Wk|Wvs|Wvc]
// epilogue: bias, signed power  sgn(t)|t|^p  for q/k, plain bf16 for v,
// scattered into per-(b,h) [N x hd] matrices for the attention GEMMs.
// ---------------------------------------------------------------------------
__global__ __launch_bounds__(256) void g1_proj(
    const u16* __restrict__ xb, const u16* __restrict__ Wcat,
    const float* __restrict__ bq, const float* __restrict__ bk,
    const float* __restrict__ bvs, const float* __restrict__ bvc,
    const float* __restrict__ pvtab,
    u16* __restrict__ Qs, u16* __restrict__ Ks, u16* __restrict__ V)
{
    __shared__ u16 As[128 * ASTR];
    __shared__ u16 Bs[64 * BSTR];
    const int bx = blockIdx.x;
    const int mt = bx / 24, nt = bx % 24;
    const int m0 = mt * 128, n0 = nt * 64;

    v8f acc[4];
    gemm_core<false>(xb, 512, Wcat, 1536, m0, n0, 512, As, Bs, acc);

    const int tid = threadIdx.x, wave = tid >> 5, lane = tid & 31, hf = lane >> 4;
#pragma unroll
    for (int ct = 0; ct < 4; ct++) {
        Acc a; a.v = acc[ct];
        const int col = n0 + ct * 16 + (lane & 15);
#pragma unroll
        for (int g = 0; g < 8; g++) {
            const int row = m0 + wave * 16 + hf * 8 + g;
            const int b = row >> 8, n = row & 255;
            const float v = a.f[g];
            if (col < 1024) {                       // q or k channel
                const int c2 = col & 511;
                const int h = c2 >> 8, d = c2 & 255;
                const float bias = (col < 512) ? bq[c2] : bk[c2];
                const float t  = v + bias;
                const float av = fabsf(t);
                float p = (av > 0.0f) ? powf(av, pvtab[c2]) : 0.0f;
                if (t < 0.0f) p = -p;               // signed power
                u16* dst = (col < 512) ? Qs : Ks;
                dst[(((size_t)b * 2 + h) * 256 + n) * 256 + d] = f2bf(p);
            } else {                                // v_s / v_c channel
                const int c2 = col - 1024;
                int h, d; float bias;
                if (c2 < 256) { h = c2 >> 7; d = c2 & 127;         bias = bvs[c2]; }
                else { const int c3 = c2 - 256; h = c3 >> 7; d = 128 + (c3 & 127); bias = bvc[c3]; }
                V[(((size_t)b * 2 + h) * 256 + n) * 256 + d] = f2bf(v + bias);
            }
        }
    }
}

// ---------------------------------------------------------------------------
// Kdenom: ksum_abs/ksum_sgn row sums + denominators
//   denom_s/c[n] = 0.5 * (|Q|^p . ksum_abs  +/-  sgn(Q)|Q|^p . ksum_sgn)
// ---------------------------------------------------------------------------
__global__ __launch_bounds__(256) void kdenom(
    const u16* __restrict__ Qs, const u16* __restrict__ Ks,
    float* __restrict__ dS, float* __restrict__ dC)
{
    __shared__ float ksa[256];
    __shared__ float kss[256];
    const int bh = blockIdx.x, t = threadIdx.x;
    const size_t off = (size_t)bh * 65536;
    {
        const uint4* kp = reinterpret_cast<const uint4*>(Ks + off + (size_t)t * 256);
        float sa = 0.0f, ss = 0.0f;
        for (int i = 0; i < 32; i++) {
            uint4 v = kp[i];
            const u16* e = reinterpret_cast<const u16*>(&v);
#pragma unroll
            for (int j = 0; j < 8; j++) { const float f = bf2f(e[j]); sa += fabsf(f); ss += f; }
        }
        ksa[t] = sa; kss[t] = ss;
    }
    __syncthreads();
    const uint4* qp = reinterpret_cast<const uint4*>(Qs + off + (size_t)t * 256);
    float a1 = 0.0f, a2 = 0.0f;
    for (int i = 0; i < 32; i++) {
        uint4 v = qp[i];
        const u16* e = reinterpret_cast<const u16*>(&v);
#pragma unroll
        for (int j = 0; j < 8; j++) {
            const float q = bf2f(e[j]);
            const int idx = i * 8 + j;
            a1 += fabsf(q) * ksa[idx];
            a2 += q * kss[idx];
        }
    }
    dS[bh * 256 + t] = 0.5f * (a1 + a2);
    dC[bh * 256 + t] = 0.5f * (a1 - a2);
}

// ---------------------------------------------------------------------------
// G2: per (b,h)  KV[256x256] = |K_sgn| @ [Vs|Vc]
// ---------------------------------------------------------------------------
__global__ __launch_bounds__(256) void g2_kv(
    const u16* __restrict__ Ks, const u16* __restrict__ V, u16* __restrict__ KV)
{
    __shared__ u16 As[128 * ASTR];
    __shared__ u16 Bs[64 * BSTR];
    const int bh = blockIdx.y;
    const int mt = blockIdx.x >> 2, nt = blockIdx.x & 3;
    const int m0 = mt * 128, n0 = nt * 64;
    const size_t off = (size_t)bh * 65536;

    v8f acc[4];
    gemm_core<true>(Ks + off, 256, V + off, 256, m0, n0, 256, As, Bs, acc);

    const int tid = threadIdx.x, wave = tid >> 5, lane = tid & 31, hf = lane >> 4;
#pragma unroll
    for (int ct = 0; ct < 4; ct++) {
        Acc a; a.v = acc[ct];
        const int col = n0 + ct * 16 + (lane & 15);
#pragma unroll
        for (int g = 0; g < 8; g++) {
            const int row = m0 + wave * 16 + hf * 8 + g;
            KV[off + (size_t)row * 256 + col] = f2bf(a.f[g]);
        }
    }
}

// ---------------------------------------------------------------------------
// G3: per (b,h)  CTX = |Q_sgn| @ KV, then /(denom+1e-6) * W_s/W_c,
// written bf16 in [B,N,C] order for the final projection.
// ---------------------------------------------------------------------------
__global__ __launch_bounds__(256) void g3_ctx(
    const u16* __restrict__ Qs, const u16* __restrict__ KV,
    const float* __restrict__ dS, const float* __restrict__ dC,
    const float* __restrict__ Ws, const float* __restrict__ Wc,
    u16* __restrict__ ctx)
{
    __shared__ u16 As[128 * ASTR];
    __shared__ u16 Bs[64 * BSTR];
    const int bh = blockIdx.y;
    const int mt = blockIdx.x >> 2, nt = blockIdx.x & 3;
    const int m0 = mt * 128, n0 = nt * 64;
    const size_t off = (size_t)bh * 65536;
    const int b = bh >> 1, h = bh & 1;

    v8f acc[4];
    gemm_core<true>(Qs + off, 256, KV + off, 256, m0, n0, 256, As, Bs, acc);

    const int tid = threadIdx.x, wave = tid >> 5, lane = tid & 31, hf = lane >> 4;
#pragma unroll
    for (int ct = 0; ct < 4; ct++) {
        Acc a; a.v = acc[ct];
        const int col = n0 + ct * 16 + (lane & 15);     // d in [0,256)
#pragma unroll
        for (int g = 0; g < 8; g++) {
            const int row = m0 + wave * 16 + hf * 8 + g;  // n in [0,256)
            const float den = ((col < 128) ? dS : dC)[bh * 256 + row] + 1e-6f;
            const float w = (col < 128) ? Ws[row * 128 + col]
                                        : Wc[row * 128 + col - 128];
            const float val = a.f[g] / den * w;
            ctx[((size_t)(b * 256 + row)) * 512 + h * 256 + col] = f2bf(val);
        }
    }
}

// ---------------------------------------------------------------------------
// G4: out[32768 x 512] = ctx @ Wp + bp   (f32 output)
// ---------------------------------------------------------------------------
__global__ __launch_bounds__(256) void g4_out(
    const u16* __restrict__ ctx, const u16* __restrict__ Wpb,
    const float* __restrict__ bp, float* __restrict__ out)
{
    __shared__ u16 As[128 * ASTR];
    __shared__ u16 Bs[64 * BSTR];
    const int mt = blockIdx.x >> 3, nt = blockIdx.x & 7;
    const int m0 = mt * 128, n0 = nt * 64;

    v8f acc[4];
    gemm_core<false>(ctx, 512, Wpb, 512, m0, n0, 512, As, Bs, acc);

    const int tid = threadIdx.x, wave = tid >> 5, lane = tid & 31, hf = lane >> 4;
#pragma unroll
    for (int ct = 0; ct < 4; ct++) {
        Acc a; a.v = acc[ct];
        const int col = n0 + ct * 16 + (lane & 15);
        const float bias = bp[col];
#pragma unroll
        for (int g = 0; g < 8; g++) {
            const int row = m0 + wave * 16 + hf * 8 + g;
            out[(size_t)row * 512 + col] = a.f[g] + bias;
        }
    }
}

// ---------------------------------------------------------------------------
extern "C" void kernel_launch(void* const* d_in, const int* in_sizes, int n_in,
                              void* d_out, int out_size, void* d_ws, size_t ws_size,
                              hipStream_t stream)
{
    const float* x    = (const float*)d_in[0];
    const float* Wq   = (const float*)d_in[1];
    const float* bq   = (const float*)d_in[2];
    const float* Wk   = (const float*)d_in[3];
    const float* bk   = (const float*)d_in[4];
    const float* Wvs  = (const float*)d_in[5];
    const float* bvs  = (const float*)d_in[6];
    const float* Wvc  = (const float*)d_in[7];
    const float* bvc  = (const float*)d_in[8];
    const float* power= (const float*)d_in[9];
    const float* Ws   = (const float*)d_in[10];
    const float* Wc   = (const float*)d_in[11];
    const float* Wp   = (const float*)d_in[12];
    const float* bp   = (const float*)d_in[13];
    float* out = (float*)d_out;

    // workspace carve-up (256B aligned), ~204 MB total
    char* p = (char*)d_ws;
    auto take = [&](size_t bytes) -> void* {
        void* r = (void*)p;
        p += (bytes + 255) & ~(size_t)255;
        return r;
    };
    u16*   xb   = (u16*)  take((size_t)32768 * 512 * 2);   // x bf16
    u16*   Wcat = (u16*)  take((size_t)512 * 1536 * 2);    // [Wq|Wk|Wvs|Wvc] bf16
    u16*   Wpb  = (u16*)  take((size_t)512 * 512 * 2);     // Wp bf16
    float* pv   = (float*)take(512 * 4);                   // 1+ALPHA*sigmoid(power)
    u16*   Qs   = (u16*)  take((size_t)256 * 65536 * 2);   // sgn(q)|q|^p per (b,h)
    u16*   Ks   = (u16*)  take((size_t)256 * 65536 * 2);   // sgn(k)|k|^p per (b,h)
    u16*   V    = (u16*)  take((size_t)256 * 65536 * 2);   // [v_s|v_c] per (b,h)
    u16*   KV   = (u16*)  take((size_t)256 * 65536 * 2);   // K_abs @ V per (b,h)
    u16*   ctx  = (u16*)  take((size_t)32768 * 512 * 2);   // scaled context, [B,N,C]
    float* dS   = (float*)take((size_t)65536 * 4);
    float* dC   = (float*)take((size_t)65536 * 4);
    (void)ws_size; (void)in_sizes; (void)n_in; (void)out_size;

    // setup
    cvt_f32_bf16<<<4096, 256, 0, stream>>>(x, xb, 32768 * 512);
    cvt_f32_bf16<<<1024, 256, 0, stream>>>(Wp, Wpb, 512 * 512);
    pack_qkv    <<<3072, 256, 0, stream>>>(Wq, Wk, Wvs, Wvc, Wcat);
    pv_kernel   <<<2,    256, 0, stream>>>(power, pv);

    // pipeline (stream order enforces dependencies)
    g1_proj<<<6144, 256, 0, stream>>>(xb, Wcat, bq, bk, bvs, bvc, pv, Qs, Ks, V);
    kdenom <<<256,  256, 0, stream>>>(Qs, Ks, dS, dC);
    g2_kv  <<<dim3(8, 256), 256, 0, stream>>>(Ks, V, KV);
    g3_ctx <<<dim3(8, 256), 256, 0, stream>>>(Qs, KV, dS, dC, Ws, Wc, ctx);
    g4_out <<<2048, 256, 0, stream>>>(ctx, Wpb, bp, out);
}
